// DFANet_34505767256766
// MI455X (gfx1250) — matrix operand order repaired and checked by reading, not compiled, not run on hardware
//
#include <hip/hip_runtime.h>
#include <hip/hip_bf16.h>
#include <climits>

// ---------------- CDNA5 vector types ----------------
typedef _Float16 v16h __attribute__((ext_vector_type(16)));
typedef _Float16 v8h  __attribute__((ext_vector_type(8)));
typedef float    v8f  __attribute__((ext_vector_type(8)));

// ---------------- problem constants ----------------
#define BATCH   2048
#define SMAX    64          // max subarrays (1024/16)
#define NMM     (4*2*SMAX)  // (z,k,s) ADC calibration entries
#define INVF    (2.0f/0.9f) // inv = 2/(1-1/ONOFF)
#define DUMMY   8.25f       // 15*(1+0.1)/2
#define SCLSUM  17.0f       // 1 + 16
#define BW      2           // b-tiles per workgroup macro-tile
#define OW      4           // o-tiles per workgroup macro-tile

// ---------------- workspace layout (bytes) ----------------
static constexpr size_t OFF_BITS = 0;
static constexpr size_t OFF_REM  = OFF_BITS + (size_t)4*BATCH*1024*2;
static constexpr size_t OFF_S0   = OFF_REM  + (size_t)2*1024*1024*2;
static constexpr size_t OFF_S0MM = OFF_S0   + (size_t)4*BATCH*SMAX*4;
static constexpr size_t OFF_PMM  = OFF_S0MM + (size_t)4*SMAX*2*4;
static constexpr size_t OFF_ROWD = OFF_PMM  + (size_t)NMM*2*4;
static constexpr size_t OFF_ACTA = OFF_ROWD + (size_t)4*BATCH*4;
static constexpr size_t OFF_ACTB = OFF_ACTA + (size_t)BATCH*1024*4;

// =====================================================================
// Bit-serial activation decomposition: q = rint(15*x); sgn/mag planes.
// =====================================================================
__global__ void cb_prep_acts(const float* __restrict__ src,
                             _Float16* __restrict__ bits,
                             int B, int cin) {
    int i = blockIdx.x * blockDim.x + threadIdx.x;
    if (i >= B * cin) return;
    float q   = rintf(15.0f * src[i]);
    float sgn = (q > 0.f) ? 1.f : ((q < 0.f) ? -1.f : 0.f);
    float mag = fabsf(q);
    size_t ps = (size_t)B * cin;
    #pragma unroll
    for (int z = 0; z < 4; ++z) {
        int m = (int)floorf(mag / (float)(1 << z));
        bits[(size_t)z * ps + i] = (_Float16)(sgn * (float)(m & 1));
    }
}

// =====================================================================
// Weight slicing: X = rint(127.5*(wq+1)); store 9*rem_k+15 as f16.
// =====================================================================
__global__ void cb_prep_w(const float* __restrict__ w,
                          _Float16* __restrict__ rem10,
                          int cout, int coutPad, int cin) {
    int i = blockIdx.x * blockDim.x + threadIdx.x;
    if (i >= coutPad * cin) return;
    int o = i / cin;
    float v0 = 0.f, v1 = 0.f;
    if (o < cout) {
        float wq = rintf(w[(size_t)o * cin + (i % cin)] * 128.f) * (1.f / 128.f);
        int X = (int)rintf(127.5f * (wq + 1.f));
        v0 = (float)(9 * (X & 15) + 15);
        v1 = (float)(9 * ((X >> 4) & 15) + 15);
    }
    size_t ss = (size_t)coutPad * cin;
    rem10[i]      = (_Float16)v0;
    rem10[ss + i] = (_Float16)v1;
}

// =====================================================================
// Re-init min/max accumulators every call.
// =====================================================================
__global__ void cb_init_mm(int* __restrict__ s0mm, int* __restrict__ pmm) {
    int i = blockIdx.x * blockDim.x + threadIdx.x;
    if (i < 4 * SMAX) { s0mm[i * 2] = INT_MAX; s0mm[i * 2 + 1] = INT_MIN; }
    if (i < NMM)      { pmm[i * 2]  = INT_MAX; pmm[i * 2 + 1]  = INT_MIN; }
}

// =====================================================================
// Dummy-column sums S0[z][b][s] + global integer min/max per (z,s).
// =====================================================================
__global__ void cb_s0(const _Float16* __restrict__ bits,
                      int* __restrict__ S0, int* __restrict__ s0mm,
                      int B, int cin, int nsub) {
    int i = blockIdx.x * blockDim.x + threadIdx.x;
    if (i >= 4 * B * nsub) return;
    int s = i % nsub, b = (i / nsub) % B, z = i / (nsub * B);
    const _Float16* p = bits + (size_t)z * B * cin + (size_t)b * cin + s * 16;
    int sum = 0;
    #pragma unroll
    for (int j = 0; j < 16; ++j) sum += (int)(float)p[j];
    S0[((size_t)z * B + b) * SMAX + s] = sum;
    atomicMin(&s0mm[(z * SMAX + s) * 2], sum);
    atomicMax(&s0mm[(z * SMAX + s) * 2 + 1], sum);
}

// =====================================================================
// Per-row dummy bias rowD[z][b] = sum_s adc_quant(8.25*S0)*17*inv
// =====================================================================
__global__ void cb_rowD(const int* __restrict__ S0, const int* __restrict__ s0mm,
                        float* __restrict__ rowD, int B, int nsub) {
    int i = blockIdx.x * blockDim.x + threadIdx.x;
    if (i >= 4 * B) return;
    int b = i % B, z = i / B;
    float sum = 0.f;
    for (int s = 0; s < nsub; ++s) {
        float dp = DUMMY * (float)S0[((size_t)z * B + b) * SMAX + s];
        float mn = DUMMY * (float)s0mm[(z * SMAX + s) * 2];
        float mx = DUMMY * (float)s0mm[(z * SMAX + s) * 2 + 1];
        float step = (mx - mn) * 0.03125f;
        float v = dp;
        if (step > 0.f) {
            float idx = fminf(fmaxf(floorf((dp - mn) / step), 0.f), 31.f);
            v = idx * step + mn;
        }
        sum += v;
    }
    rowD[(size_t)z * B + b] = sum * SCLSUM * INVF;
}

// =====================================================================
// Core crossbar GEMM. 8 waves per workgroup; each wave owns a 16x16
// (b,o) tile of a 32x64 macro-tile; loops z(4) x s(nsub) x k(2); one
// v_wmma_f32_16x16x32_f16 per step (subarray in K=0..15, zeros above;
// exact integer math: part*10 = 9*S1 + 15*S0). The s-loop is unrolled
// x2 with ping-pong PERSISTENT operand buffers: A's upper-K zeros and
// B's upper-K zeros are zeroed once per z-pass (loop-invariant regs);
// loads overwrite only the live halves, so no
// per-iteration zero/rotation movs, and next-subarray loads stay in
// flight across each WMMA pair.
//   MODE 0: float-domain min/max tile reduce (exact ints), then one
//           lane-distinct ds_min_i32/ds_max_i32 into a [k][s][lane]
//           LDS table (conflict-free, no wait); per-z 32->1 flush.
//   MODE 1: branchless 6-op/element ADC dequant. Tables hold
//           (mn, step, istep, moff) with istep=moff=0 when step==0,
//           which makes the formula yield mn == the passthrough value
//           (min==max => part==mn), so no select is needed.
// =====================================================================
template <int MODE>
__global__ __launch_bounds__(256) void cb_gemm(
        const _Float16* __restrict__ bits,
        const _Float16* __restrict__ rem10,
        int* __restrict__ pmm,
        const float* __restrict__ rowD,
        float* __restrict__ dst,
        int B, int cin, int cout, int coutPad, int nsub,
        float outScale, int do_tanh, int dstStride) {
    __shared__ int   lds_min[2 * SMAX * 32];   // [k][s][lane], per-z reuse
    __shared__ int   lds_max[2 * SMAX * 32];
    __shared__ float lds_mn[NMM];
    __shared__ float lds_step[NMM];
    __shared__ float lds_istep[NMM];
    __shared__ float lds_moff[NMM];

    const int tid  = threadIdx.x;
    const int wave = tid >> 5;
    const int lane = tid & 31;
    const int hi   = lane >> 4;

    if (MODE == 1) {
        for (int i = tid; i < NMM; i += 256) {
            float mn10 = (float)pmm[i * 2];
            float mx10 = (float)pmm[i * 2 + 1];
            float st10 = (mx10 - mn10) * 0.03125f;          // ADC 2^-5, x10 dom
            float ist  = (st10 > 0.f) ? (1.f / st10) : 0.f; // div off hot path
            lds_mn[i]    = 0.1f * mn10;
            lds_step[i]  = 0.1f * st10;
            lds_istep[i] = ist;
            lds_moff[i]  = -mn10 * ist;
        }
        __syncthreads();
    }

    const int btile = blockIdx.x * BW + (wave & (BW - 1));
    int otile = blockIdx.y * OW + (wave >> 1);
    const int maxOt = coutPad / 16 - 1;
    if (otile > maxOt) otile = maxOt;          // clamp: duplicates are benign
    const int bbase = btile * 16;
    const int arow  = bbase + (lane & 15);     // A-operand source row
    const int o     = otile * 16 + (lane & 15);
    const size_t planeStride = (size_t)B * cin;
    const _Float16* sl0 = rem10;
    const _Float16* sl1 = rem10 + (size_t)coutPad * cin;

    float acc[8];
    #pragma unroll
    for (int r = 0; r < 8; ++r) acc[r] = 0.f;

    for (int z = 0; z < 4; ++z) {
        const _Float16* bz = bits + (size_t)z * planeStride;

        if (MODE == 0) {                       // per-z table (re)init
            for (int i = tid; i < 2 * SMAX * 32; i += 256) {
                lds_min[i] = INT_MAX;
                lds_max[i] = INT_MIN;
            }
            __syncthreads();
        }

        float accz[8];
        #pragma unroll
        for (int r = 0; r < 8; ++r) accz[r] = 0.f;

        // Persistent operand buffers: zeros are loop-invariant registers.
        v16h aFA = {}, aFB = {};               // upper 8 halves stay zero
        v16h b0A = {}, b1A = {}, b0B = {}, b1B = {};  // upper 8 halves stay zero

        auto ldA = [&](v16h& dv, int s) {      // low half only
            v8h t = *reinterpret_cast<const v8h*>(
                bz + (size_t)arow * cin + s * 16 + hi * 8);
            #pragma unroll
            for (int j = 0; j < 8; ++j) dv[j] = t[j];
        };
        auto ldB = [&](v16h& dv, const _Float16* slice, int s) {
            v8h t = *reinterpret_cast<const v8h*>(
                slice + (size_t)o * cin + s * 16 + hi * 8);
            #pragma unroll
            for (int j = 0; j < 8; ++j) dv[j] = t[j];
        };

        // Per-subarray compute: 2 WMMAs (weight slices) + epilogue.
        auto compute = [&](int s, const v16h& a, const v16h& b0, const v16h& b1) {
            #pragma unroll
            for (int k = 0; k < 2; ++k) {
                v8f c = {};
                c = __builtin_amdgcn_wmma_f32_16x16x32_f16(
                        false, a, false, (k == 0) ? b0 : b1,
                        (short)0, c, false, false);
                if (MODE == 0) {
                    float fmn = fminf(c[0], c[1]);
                    float fmx = fmaxf(c[0], c[1]);
                    #pragma unroll
                    for (int r = 2; r < 8; ++r) {
                        fmn = fminf(fmn, c[r]);
                        fmx = fmaxf(fmx, c[r]);
                    }
                    int lmin = (int)fmn, lmax = (int)fmx;
                    if (o >= cout) { lmin = INT_MAX; lmax = INT_MIN; }
                    const int off = (k * SMAX + s) * 32 + lane;   // lane-distinct
                    atomicMin(&lds_min[off], lmin);   // vector ds_min, no wait
                    atomicMax(&lds_max[off], lmax);   // vector ds_max, no wait
                } else {
                    const int idx = (z * 2 + k) * SMAX + s;
                    const float mn    = lds_mn[idx];
                    const float step  = lds_step[idx];
                    const float istep = lds_istep[idx];
                    const float moff  = lds_moff[idx];
                    const float wk = (k == 0) ? 1.f : 16.f;
                    #pragma unroll
                    for (int r = 0; r < 8; ++r) {
                        float qi = floorf(fmaf(c[r], istep, moff));
                        qi = fminf(fmaxf(qi, 0.f), 31.f);
                        accz[r] = fmaf(fmaf(qi, step, mn), wk, accz[r]);
                    }
                }
            }
        };

        // ---- ping-pong pipelined s-loop (unroll x2) ----
        ldA(aFA, 0); ldB(b0A, sl0, 0); ldB(b1A, sl1, 0);
        int s = 0;
        while (true) {
            int sn = (s + 1 < nsub) ? (s + 1) : s;
            ldA(aFB, sn); ldB(b0B, sl0, sn); ldB(b1B, sl1, sn);
            compute(s, aFA, b0A, b1A);
            if (++s >= nsub) break;

            sn = (s + 1 < nsub) ? (s + 1) : s;
            ldA(aFA, sn); ldB(b0A, sl0, sn); ldB(b1A, sl1, sn);
            compute(s, aFB, b0B, b1B);
            if (++s >= nsub) break;
        }

        if (MODE == 0) {                       // per-z flush: 32->1 + global
            __syncthreads();
            for (int e = tid; e < 2 * SMAX; e += 256) {
                int mn = INT_MAX, mx = INT_MIN;
                #pragma unroll 8
                for (int j = 0; j < 32; ++j) {
                    mn = min(mn, lds_min[e * 32 + j]);
                    mx = max(mx, lds_max[e * 32 + j]);
                }
                const int gidx = (z * 2 + (e / SMAX)) * SMAX + (e % SMAX);
                atomicMin(&pmm[gidx * 2], mn);
                atomicMax(&pmm[gidx * 2 + 1], mx);
            }
            __syncthreads();                   // before next-z re-init
        } else {
            float zw = (float)(1 << z);
            #pragma unroll
            for (int r = 0; r < 8; ++r) {
                int b = bbase + r + hi * 8;    // C/D layout: VGPR r, hi -> M=r+8
                acc[r] += (accz[r] * INVF - rowD[(size_t)z * B + b]) * zw;
            }
        }
    }

    if (MODE == 1 && o < cout) {
        #pragma unroll
        for (int r = 0; r < 8; ++r) {
            int b = bbase + r + hi * 8;
            float v = acc[r] * outScale;                 // /2^4 /2^8 /scale
            v = fminf(fmaxf(v, -0.875f), 0.875f);        // WAGE clip
            v = rintf(v * 8.f) * 0.125f;                 // WAGE 4-bit quant
            if (do_tanh) v = tanhf(v);
            dst[(size_t)b * dstStride + o] = v;
        }
    }
}

// =====================================================================
// Host-side per-layer driver (all launches on `stream`; no allocs).
// =====================================================================
static void run_layer(const float* act, const float* w, float* dst,
                      int B, int cin, int cout, int coutPad, float scale,
                      int do_tanh, int dstStride, unsigned char* ws,
                      hipStream_t stream) {
    _Float16* bits  = (_Float16*)(ws + OFF_BITS);
    _Float16* rem10 = (_Float16*)(ws + OFF_REM);
    int*      S0    = (int*)(ws + OFF_S0);
    int*      s0mm  = (int*)(ws + OFF_S0MM);
    int*      pmm   = (int*)(ws + OFF_PMM);
    float*    rowD  = (float*)(ws + OFF_ROWD);
    const int nsub  = cin / 16;

    int n;
    n = B * cin;
    cb_prep_acts<<<(n + 255) / 256, 256, 0, stream>>>(act, bits, B, cin);
    n = coutPad * cin;
    cb_prep_w<<<(n + 255) / 256, 256, 0, stream>>>(w, rem10, cout, coutPad, cin);
    cb_init_mm<<<(NMM + 255) / 256, 256, 0, stream>>>(s0mm, pmm);
    n = 4 * B * nsub;
    cb_s0<<<(n + 255) / 256, 256, 0, stream>>>(bits, S0, s0mm, B, cin, nsub);

    dim3 grid(B / (16 * BW), (coutPad / 16 + OW - 1) / OW);
    cb_gemm<0><<<grid, 256, 0, stream>>>(bits, rem10, pmm, rowD, dst,
                                         B, cin, cout, coutPad, nsub,
                                         0.f, 0, dstStride);
    n = 4 * B;
    cb_rowD<<<(n + 255) / 256, 256, 0, stream>>>(S0, s0mm, rowD, B, nsub);

    float outScale = 1.f / (16.f * 256.f * scale);
    cb_gemm<1><<<grid, 256, 0, stream>>>(bits, rem10, pmm, rowD, dst,
                                         B, cin, cout, coutPad, nsub,
                                         outScale, do_tanh, dstStride);
}

extern "C" void kernel_launch(void* const* d_in, const int* in_sizes, int n_in,
                              void* d_out, int out_size, void* d_ws, size_t ws_size,
                              hipStream_t stream) {
    (void)in_sizes; (void)n_in; (void)out_size; (void)ws_size;
    const float* x  = (const float*)d_in[0];   // [2048, 784]
    const float* w1 = (const float*)d_in[1];   // [512, 784]
    const float* w2 = (const float*)d_in[2];   // [1024, 512]
    const float* w3 = (const float*)d_in[3];   // [10, 1024]
    unsigned char* ws = (unsigned char*)d_ws;
    float* actA = (float*)(ws + OFF_ACTA);
    float* actB = (float*)(ws + OFF_ACTB);

    // WAGE scales: 784->16, 512->8, 1024->16
    run_layer(x,    w1, actA,          BATCH,  784,  512,  512, 16.f, 1,  512, ws, stream);
    run_layer(actA, w2, actB,          BATCH,  512, 1024, 1024,  8.f, 1, 1024, ws, stream);
    run_layer(actB, w3, (float*)d_out, BATCH, 1024,   10,   16, 16.f, 0,   10, ws, stream);
}
